// Attention_28587302322787
// MI455X (gfx1250) — matrix operand order repaired and checked by reading, not compiled, not run on hardware
//
#include <hip/hip_runtime.h>

// ---------------------------------------------------------------------------
// Causal flash attention for gfx1250 (CDNA5), wave32, WMMA f16 path.
// B=4, L=2048, H=16, E=64, fp32 in/out, f16 matmul operands, fp32 accum.
// Per 32-key step, per wave: 4 WMMA (Q*K^T) + 4 WMMA (P*V) + 1 WMMA (rowsum).
// ---------------------------------------------------------------------------

typedef __attribute__((ext_vector_type(16))) _Float16 v16h;
typedef __attribute__((ext_vector_type(2)))  __fp16   h2v;   // cvt_pkrtz result type
typedef __attribute__((ext_vector_type(8)))  float    v8f;

#define B_    4
#define L_    2048
#define H_    16
#define E_    64

#define QROWS 16            // query rows per wave (WMMA M)
#define WAVES 8             // waves per block
#define QBLK  (QROWS*WAVES) // 128 query rows per block
#define SBLK  32            // key tile (WMMA K for PV, 2x16 N for QK^T)

// LDS strides in f16 elements.
// KSTR/PSTR: rows 16B-aligned so fragments load as ds_load_b128 (bank sweeps
// 36n / 20n mod 64 stay conflict-free). VSTR stays 2*odd so the transposed
// staging writes are conflict-free (reads use b32 pairs there).
#define KSTR  72            // K tile row-major [32][64], 144 B rows
#define VSTR  34            // V tile transposed [64][32], 68 B rows
#define PSTR  40            // per-wave P tile [16][32], 80 B rows

// hardware exp2: bare v_exp_f32
__device__ __forceinline__ float hexp2(float x) {
    return __builtin_amdgcn_exp2f(x);
}

// max-reduce across the 16-lane half (xor 8,4,2,1) via ds_swizzle (no addr setup)
__device__ __forceinline__ float hmax16(float x) {
    x = fmaxf(x, __int_as_float(__builtin_amdgcn_ds_swizzle(__float_as_int(x), 0x201f)));
    x = fmaxf(x, __int_as_float(__builtin_amdgcn_ds_swizzle(__float_as_int(x), 0x101f)));
    x = fmaxf(x, __int_as_float(__builtin_amdgcn_ds_swizzle(__float_as_int(x), 0x081f)));
    x = fmaxf(x, __int_as_float(__builtin_amdgcn_ds_swizzle(__float_as_int(x), 0x041f)));
    return x;
}

__global__ __launch_bounds__(256, 2)
void fa_causal_wmma(const float* __restrict__ Q,
                    const float* __restrict__ K,
                    const float* __restrict__ V,
                    float* __restrict__ O) {
    __shared__ __align__(16) _Float16 ldsK[SBLK * KSTR];        // [s][e]
    __shared__ __align__(16) _Float16 ldsV[E_ * VSTR];          // [e][s] transposed
    __shared__ __align__(16) _Float16 ldsP[WAVES * QROWS * PSTR];

    const int tid  = threadIdx.x;
    const int wave = tid >> 5;
    const int lane = tid & 31;
    const int hh   = lane >> 4;       // lane half: 0 or 1
    const int l16  = lane & 15;

    // grid = B * H * (L/QBLK)
    int blk = blockIdx.x;
    const int qt = blk % (L_ / QBLK); blk /= (L_ / QBLK);
    const int h  = blk % H_;          blk /= H_;
    const int b  = blk;

    const int    qbase  = qt * QBLK + wave * QROWS;
    const size_t bh_off = (size_t)b * L_ * (H_ * E_) + (size_t)h * E_;
    const int    rowstr = H_ * E_;

    // scale folded into Q, in base-2: exp(s/8) == exp2(s * 0.125 * log2(e))
    const float qscale = 0.125f * 1.44269504088896340736f;

    // ---- Q A-fragments (16 rows x 64, two k=32 slices), f16, pre-scaled ----
    // A 16x32 layout: lane<16: m=l16, K(j) = (j<4 ? 2j : 16+2(j-4));
    //                 lane>=16: same rows, K offset +8.
    v16h qa[2];
    {
        const float* qrow = Q + bh_off + (size_t)(qbase + l16) * rowstr + 8 * hh;
#pragma unroll
        for (int f = 0; f < 2; ++f) {
            const float4 a  = *(const float4*)(qrow + 32 * f);      // j=0,1
            const float4 bq = *(const float4*)(qrow + 32 * f + 4);  // j=2,3
            const float4 c  = *(const float4*)(qrow + 32 * f + 16); // j=4,5
            const float4 d  = *(const float4*)(qrow + 32 * f + 20); // j=6,7
            union { v16h v; h2v h[8]; } u;
            u.h[0] = __builtin_amdgcn_cvt_pkrtz(a.x * qscale, a.y * qscale);
            u.h[1] = __builtin_amdgcn_cvt_pkrtz(a.z * qscale, a.w * qscale);
            u.h[2] = __builtin_amdgcn_cvt_pkrtz(bq.x * qscale, bq.y * qscale);
            u.h[3] = __builtin_amdgcn_cvt_pkrtz(bq.z * qscale, bq.w * qscale);
            u.h[4] = __builtin_amdgcn_cvt_pkrtz(c.x * qscale, c.y * qscale);
            u.h[5] = __builtin_amdgcn_cvt_pkrtz(c.z * qscale, c.w * qscale);
            u.h[6] = __builtin_amdgcn_cvt_pkrtz(d.x * qscale, d.y * qscale);
            u.h[7] = __builtin_amdgcn_cvt_pkrtz(d.z * qscale, d.w * qscale);
            qa[f] = u.v;
        }
    }

    // constant all-ones B fragment: row-sums of P via one extra WMMA
    v16h onesv;
    {
        union { v16h v; unsigned short u[16]; } c;
#pragma unroll
        for (int i = 0; i < 16; ++i) c.u[i] = 0x3C00;   // f16 1.0
        onesv = c.v;
    }

    // ---- accumulators -------------------------------------------------------
    v8f o[4] = {v8f{}, v8f{}, v8f{}, v8f{}};   // 16x64 output, fp32
    v8f lacc = {};                             // per-row softmax denominators
    float mgrp = -1e30f;                       // running max of the 8-row group

    const int nsteps = (qt * QBLK + QBLK) / SBLK;     // block-uniform
    const int qmax_u = __builtin_amdgcn_readfirstlane(qbase + QROWS - 1);
    const int qb_u   = __builtin_amdgcn_readfirstlane(qbase);

    // ---- strength-reduced staging pointers (advance by SBLK rows/step) -----
    const int s_t = tid >> 4;
    const int e_t = (tid & 15) * 4;
    const float* kp = K + bh_off + (size_t)s_t * rowstr + e_t;
    const float* vp = V + bh_off + (size_t)s_t * rowstr + e_t;
    _Float16* kw = &ldsK[s_t * KSTR + e_t];
    _Float16* vw = &ldsV[e_t * VSTR + s_t];
    const int gstep = 16 * rowstr;            // second staging row offset

    for (int step = 0; step < nsteps; ++step) {
        const int s0 = step * SBLK;

        __syncthreads();   // previous iteration done with ldsK/ldsV

        // ---- cooperative K/V tile staging: 32 x 64 fp32 -> f16 --------------
#pragma unroll
        for (int it = 0; it < 2; ++it) {
            const float4 kv = *(const float4*)(kp + it * gstep);
            const float4 vv = *(const float4*)(vp + it * gstep);
            h2v* kd = (h2v*)(kw + it * 16 * KSTR);
            kd[0] = __builtin_amdgcn_cvt_pkrtz(kv.x, kv.y);
            kd[1] = __builtin_amdgcn_cvt_pkrtz(kv.z, kv.w);
            _Float16* vd = vw + it * 16;
            vd[0 * VSTR] = (_Float16)vv.x;
            vd[1 * VSTR] = (_Float16)vv.y;
            vd[2 * VSTR] = (_Float16)vv.z;
            vd[3 * VSTR] = (_Float16)vv.w;
        }
        kp += SBLK * rowstr;
        vp += SBLK * rowstr;
        __syncthreads();

        // prefetch next K/V tile into cache while this tile computes
        if (step + 1 < nsteps) {
#pragma unroll
            for (int it = 0; it < 2; ++it) {
                __builtin_prefetch(kp + it * gstep, 0, 3);
                __builtin_prefetch(vp + it * gstep, 0, 3);
            }
        }

        if (s0 <= qmax_u) {   // wave-uniform causal skip (barriers stay outside)
            // ---- S = Q * K^T : 16x32 scores, 4x wmma_f32_16x16x32_f16 -------
            v8f c0 = {}, c1 = {};
#pragma unroll
            for (int f = 0; f < 2; ++f) {       // E slices 0..31, 32..63
                union { v16h v; uint4 q[2]; } b0, b1;
                const _Float16* k0 = &ldsK[(l16)      * KSTR + 8 * hh + 32 * f];
                const _Float16* k1 = &ldsK[(16 + l16) * KSTR + 8 * hh + 32 * f];
                b0.q[0] = *(const uint4*)(k0);
                b0.q[1] = *(const uint4*)(k0 + 16);
                b1.q[0] = *(const uint4*)(k1);
                b1.q[1] = *(const uint4*)(k1 + 16);
                c0 = __builtin_amdgcn_wmma_f32_16x16x32_f16(
                         false, qa[f], false, b0.v, (short)0, c0, false, false);
                c1 = __builtin_amdgcn_wmma_f32_16x16x32_f16(
                         false, qa[f], false, b1.v, (short)0, c1, false, false);
            }

            // ---- online softmax (C layout: row m=r+8*hh, col n=l16/+16) -----
            if (s0 + SBLK - 1 > qb_u) {        // only the diagonal tile masks
#pragma unroll
                for (int r = 0; r < 8; ++r) {
                    const int qrow = qbase + r + 8 * hh;
                    if (s0 + l16      > qrow) c0[r] = -1e30f;
                    if (s0 + 16 + l16 > qrow) c1[r] = -1e30f;
                }
            }

            // group max: in-lane over 8 rows x 2 cols, then 16-lane reduce
            float vmax = fmaxf(c0[0], c1[0]);
#pragma unroll
            for (int r = 1; r < 8; ++r) vmax = fmaxf(vmax, fmaxf(c0[r], c1[r]));
            vmax = hmax16(vmax);

            // rescale running state only when the running max grows
#if __has_builtin(__builtin_amdgcn_ballot_w32)
            const bool grew = __builtin_amdgcn_ballot_w32(vmax > mgrp) != 0u;
#else
            const bool grew = true;
#endif
            if (grew) {
                const float mnew = fmaxf(mgrp, vmax);
                const float corr = hexp2(mgrp - mnew);
                mgrp = mnew;
#pragma unroll
                for (int r = 0; r < 8; ++r) {
                    o[0][r] *= corr; o[1][r] *= corr;
                    o[2][r] *= corr; o[3][r] *= corr;
                    lacc[r] *= corr;
                }
            }

            // p = exp2(x - m), stash as f16 in per-wave LDS tile
#pragma unroll
            for (int r = 0; r < 8; ++r) {
                _Float16* pp = &ldsP[(wave * QROWS + r + 8 * hh) * PSTR];
                pp[l16]      = (_Float16)hexp2(c0[r] - mgrp);
                pp[16 + l16] = (_Float16)hexp2(c1[r] - mgrp);
            }

            // cross-lane LDS visibility within the wave (CDNA5 split counter)
            asm volatile("s_wait_dscnt 0x0" ::: "memory");

            // ---- P in A-fragment layout (16x32), two b128 loads -------------
            union { v16h v; uint4 q[2]; } pa;
            const _Float16* pb = &ldsP[(wave * QROWS + l16) * PSTR + 8 * hh];
            pa.q[0] = *(const uint4*)(pb);
            pa.q[1] = *(const uint4*)(pb + 16);

            // row-sums: l += P * ones  (replicated across the 16-lane half)
            lacc = __builtin_amdgcn_wmma_f32_16x16x32_f16(
                       false, pa.v, false, onesv, (short)0, lacc, false, false);

            // ---- O += P * V : 4x wmma over E tiles of 16 --------------------
#pragma unroll
            for (int t = 0; t < 4; ++t) {
                union { v16h v; unsigned p[8]; } bv;
#pragma unroll
                for (int j = 0; j < 8; ++j) {
                    const int k = 2 * j + 8 * hh + 8 * (j >> 2); // K = s offset
                    bv.p[j] = *(const unsigned*)&ldsV[(t * 16 + l16) * VSTR + k];
                }
                o[t] = __builtin_amdgcn_wmma_f32_16x16x32_f16(
                           false, pa.v, false, bv.v, (short)0, o[t], false, false);
            }
        }
    }

    // ---- normalize and store ------------------------------------------------
    float inv[8];
#pragma unroll
    for (int r = 0; r < 8; ++r) inv[r] = 1.0f / lacc[r];
#pragma unroll
    for (int t = 0; t < 4; ++t) {
#pragma unroll
        for (int r = 0; r < 8; ++r) {
            const int m = r + 8 * hh;
            O[bh_off + (size_t)(qbase + m) * rowstr + t * 16 + l16] = o[t][r] * inv[r];
        }
    }
}

extern "C" void kernel_launch(void* const* d_in, const int* in_sizes, int n_in,
                              void* d_out, int out_size, void* d_ws, size_t ws_size,
                              hipStream_t stream) {
    (void)in_sizes; (void)n_in; (void)out_size; (void)d_ws; (void)ws_size;
    const float* Q = (const float*)d_in[0];
    const float* K = (const float*)d_in[1];
    const float* V = (const float*)d_in[2];
    float*       O = (float*)d_out;

    const int nblocks = B_ * H_ * (L_ / QBLK);   // 1024 blocks x 256 threads
    fa_causal_wmma<<<nblocks, 256, 0, stream>>>(Q, K, V, O);
}
